// CrossHeadAttention_592705487072
// MI455X (gfx1250) — matrix operand order repaired and checked by reading, not compiled, not run on hardware
//
#include <hip/hip_runtime.h>

#define EMB   768
#define NH    8
#define HD    96
#define BATCH 2
#define SEQ   2048
#define MROWS (BATCH*SEQ)   /* 4096 */
#define NQKV  (3*EMB)       /* 2304 */
#define SCALE_F 27.712812921102035f
#define LN_EPS  1e-5f

typedef __bf16 bf16_t;
typedef __attribute__((ext_vector_type(16))) __bf16 v16bf;
typedef __attribute__((ext_vector_type(8)))  __bf16 v8bf;
typedef __attribute__((ext_vector_type(8)))  float  v8f;

union BF16x16 { v16bf v; v8bf h[2]; };

// ---------------------------------------------------------------------------
// fp32 -> bf16 elementwise conversion
// ---------------------------------------------------------------------------
__global__ __launch_bounds__(256)
void cvt_f32_bf16_kernel(const float* __restrict__ src, bf16_t* __restrict__ dst, int n) {
    int i = blockIdx.x * 256 + threadIdx.x;
    if (i < n) dst[i] = (bf16_t)src[i];
}

// ---------------------------------------------------------------------------
// GEMM: C[M,Nc] = A[M,K](bf16) x B[K,Nc](bf16) + bias  (f32 out)
// block = 256 threads = 8 waves; wave tile = 16x64; block tile = 64x128.
// grid = (M/64, Nc/128).  K/32 must be even; Nc % 128 == 0; M % 64 == 0.
// Double-buffered and manually 2x unrolled so the buffers ping-pong without
// register copies; next chunk's loads issue before current chunk's WMMAs.
// waves_per_eu(2) gives the RA enough VGPRs to keep both buffers resident
// (2 waves/SIMD is all that's needed to host one 8-wave workgroup per WGP).
// ---------------------------------------------------------------------------
__global__ __launch_bounds__(256)
__attribute__((amdgpu_waves_per_eu(2)))
void gemm_bf16_wmma_kernel(const bf16_t* __restrict__ A, const bf16_t* __restrict__ Bw,
                           const float* __restrict__ bias, float* __restrict__ C,
                           int K, int Nc) {
    const int lane = threadIdx.x & 31;
    const int wave = threadIdx.x >> 5;
    const int rg = wave & 3;            // 4 row groups
    const int cg = wave >> 2;           // 2 col groups
    const int m0 = blockIdx.x * 64 + rg * 16;
    const int n0 = blockIdx.y * 128 + cg * 64;

    const int mA = m0 + (lane & 15);
    const int kb = (lane >> 4) * 8;     // A-fragment K sub-offset per half-wave

    v8f zero = {0.f,0.f,0.f,0.f,0.f,0.f,0.f,0.f};
    v8f acc[4];
#pragma unroll
    for (int t = 0; t < 4; ++t) acc[t] = zero;

    const bf16_t* ap = A + (size_t)mA * K + kb;          // += 32 per chunk
    const bf16_t* bp = Bw + n0 + (lane & 15);            // += 32*Nc per chunk
    const size_t bstep = (size_t)32 * Nc;

    BF16x16 a0, a1;
    v16bf b0[4], b1[4];

    auto loadA = [&](BF16x16& d) {
        d.h[0] = *(const v8bf*)(ap);
        d.h[1] = *(const v8bf*)(ap + 16);
        ap += 32;
    };
    auto loadB = [&](v16bf (&d)[4]) {
#pragma unroll
        for (int t = 0; t < 4; ++t)
#pragma unroll
            for (int s = 0; s < 16; ++s)
                d[t][s] = bp[(size_t)(kb + (s & 7) + (s >> 3) * 16) * Nc + 16 * t];
        bp += bstep;
    };
    auto mma = [&](BF16x16& a, v16bf (&bb)[4]) {
#pragma unroll
        for (int t = 0; t < 4; ++t)
            acc[t] = __builtin_amdgcn_wmma_f32_16x16x32_bf16(
                false, a.v, false, bb[t], (short)0, acc[t], false, false);
    };

    const int nch = K >> 5;             // number of 32-wide chunks (even)
    loadA(a0); loadB(b0);
    for (int ch = 0; ch + 2 < nch; ch += 2) {
        loadA(a1); loadB(b1);
        mma(a0, b0);
        loadA(a0); loadB(b0);
        mma(a1, b1);
    }
    loadA(a1); loadB(b1);
    mma(a0, b0);
    mma(a1, b1);

    const int hi = lane >> 4, col0 = lane & 15;
#pragma unroll
    for (int t = 0; t < 4; ++t) {
        int n = n0 + 16 * t + col0;
        float bv = bias[n];
#pragma unroll
        for (int r = 0; r < 8; ++r) {
            int m = m0 + r + 8 * hi;
            C[(size_t)m * Nc + n] = acc[t][r] + bv;
        }
    }
}

// ---------------------------------------------------------------------------
// Split interleaved QKV [M, 2304] (col = h*288 + d*3 + {0,1,2}) into
// Q  [bh, n, d]  bf16
// Kt [bh, d, n]  bf16   (pre-transposed for the QK^T B-fragment)
// V  [bh, n, d]  bf16
// ---------------------------------------------------------------------------
__global__ __launch_bounds__(256)
void split_qkv_kernel(const float* __restrict__ qkv,
                      bf16_t* __restrict__ Q, bf16_t* __restrict__ Kt,
                      bf16_t* __restrict__ V, int wq, int wk, int wv) {
    int idx = blockIdx.x * 256 + threadIdx.x;      // over MROWS*EMB
    if (idx >= MROWS * EMB) return;
    int row = idx / EMB;                            // b*SEQ + n
    int c   = idx - row * EMB;                      // h*HD + d
    int h = c / HD, d = c - h * HD;
    int b = row / SEQ, n = row - b * SEQ;
    const float* p = qkv + (size_t)row * NQKV + h * (HD * 3) + d * 3;
    size_t bh = (size_t)(b * NH + h);
    if (wq) Q [(bh * SEQ + n) * HD + d] = (bf16_t)p[0];
    if (wk) Kt[(bh * HD + d) * SEQ + n] = (bf16_t)p[1];
    if (wv) V [(bh * SEQ + n) * HD + d] = (bf16_t)p[2];
}

// ---------------------------------------------------------------------------
// Flash attention (softmax first, THEN /sqrt(EMB), per reference).
// block = 256 = 8 waves; each wave owns 16 queries of one (b,h).
// grid = B*H*(SEQ/128).  Output bf16, layout [b, n, h*HD + d] (= [M, EMB]).
// Double-buffered K/V fragments: each step prefetches the NEXT step's 12
// fragments first, so their latency hides behind this step's WMMAs + softmax.
// Runs one phantom prefetch past the end => Kt/V buffers carry 8KB padding.
// ---------------------------------------------------------------------------
struct WaveSmem {
    float   S[16][33];     // raw energies for 16 queries x 32 keys
    bf16_t  P[16][32];     // exp(S - max) as bf16 (A-fragment source)
    float   scl[16];       // per-row rescale of accumulators
    float   sum[16];       // final softmax denominators
};

__device__ __attribute__((always_inline))
static void attn_step(WaveSmem& sm, v8f (&acc)[6], const BF16x16 (&aq)[3],
                      v16bf (&kfc)[6], v16bf (&vfc)[6],
                      v16bf (&kfn)[6], v16bf (&vfn)[6],
                      const bf16_t*& vp, const bf16_t*& kp,
                      float& run_max, float& run_sum,
                      int lane, int m16, int hi) {
    v8f zero = {0.f,0.f,0.f,0.f,0.f,0.f,0.f,0.f};

    // ---- prefetch next step's V and K fragments (distinct registers) ----
#pragma unroll
    for (int t6 = 0; t6 < 6; ++t6)
#pragma unroll
        for (int s = 0; s < 16; ++s)
            vfn[t6][s] = vp[(size_t)(hi * 8 + (s & 7) + (s >> 3) * 16) * HD + 16 * t6];
#pragma unroll
    for (int t = 0; t < 2; ++t)
#pragma unroll
        for (int c = 0; c < 3; ++c)
#pragma unroll
            for (int s = 0; s < 16; ++s)
                kfn[t * 3 + c][s] = kp[(size_t)(c * 32 + hi * 8 + (s & 7) + (s >> 3) * 16) * SEQ + 16 * t];
    vp += (size_t)32 * HD;
    kp += 32;

    // ---- S = Q x K^T for 32 keys (two 16x16 tiles, 3 WMMAs each) ----
#pragma unroll
    for (int t = 0; t < 2; ++t) {
        v8f s = zero;
#pragma unroll
        for (int c = 0; c < 3; ++c)
            s = __builtin_amdgcn_wmma_f32_16x16x32_bf16(
                false, aq[c].v, false, kfc[t * 3 + c], (short)0, s, false, false);
#pragma unroll
        for (int r = 0; r < 8; ++r)
            sm.S[r + 8 * hi][16 * t + m16] = s[r];
    }
    __syncthreads();

    // ---- online softmax stats: lane m (< 16) owns query row m ----
    if (lane < 16) {
        float mx = run_max;
#pragma unroll
        for (int kk = 0; kk < 32; ++kk) mx = fmaxf(mx, sm.S[lane][kk]);
        float scale = __expf(run_max - mx);
        float ps = 0.f;
#pragma unroll
        for (int kk = 0; kk < 32; ++kk) {
            float p = __expf(sm.S[lane][kk] - mx);
            sm.P[lane][kk] = (bf16_t)p;
            ps += p;
        }
        run_sum = run_sum * scale + ps;
        run_max = mx;
        sm.scl[lane] = scale;
    }
    __syncthreads();

    // ---- rescale accumulators by per-row scale ----
    float sc[8];
#pragma unroll
    for (int r = 0; r < 8; ++r) sc[r] = sm.scl[r + 8 * hi];
#pragma unroll
    for (int t6 = 0; t6 < 6; ++t6)
#pragma unroll
        for (int r = 0; r < 8; ++r) acc[t6][r] *= sc[r];

    // ---- regather P into a bf16 A-fragment (16 queries x 32 keys) ----
    v16bf pa;
#pragma unroll
    for (int e = 0; e < 16; ++e)
        pa[e] = sm.P[m16][hi * 8 + (e >> 3) * 16 + (e & 7)];

    // ---- acc += P x V (6 WMMAs, V fragments were prefetched last step) ----
#pragma unroll
    for (int t6 = 0; t6 < 6; ++t6)
        acc[t6] = __builtin_amdgcn_wmma_f32_16x16x32_bf16(
            false, pa, false, vfc[t6], (short)0, acc[t6], false, false);
    __syncthreads();
}

__global__ __launch_bounds__(256)
__attribute__((amdgpu_waves_per_eu(2)))
void attention_kernel(const bf16_t* __restrict__ Q, const bf16_t* __restrict__ Kt,
                      const bf16_t* __restrict__ V, bf16_t* __restrict__ Out) {
    __shared__ WaveSmem sm_all[8];
    const int lane = threadIdx.x & 31;
    const int wave = threadIdx.x >> 5;
    WaveSmem& sm = sm_all[wave];

    const int qc = blockIdx.x & (SEQ / 128 - 1);   // query chunk (16 per bh)
    const int bh = blockIdx.x >> 4;                // 0..B*H-1
    const int b  = bh >> 3, h = bh & 7;
    const int q0 = qc * 128 + wave * 16;

    const bf16_t* Qb = Q  + ((size_t)bh * SEQ + q0) * HD;
    const bf16_t* Kb = Kt + (size_t)bh * HD * SEQ;
    const bf16_t* Vb = V  + (size_t)bh * SEQ * HD;

    const int m16 = lane & 15, hi = lane >> 4;

    // Q fragments for the 3 K-chunks of the D=96 contraction
    BF16x16 aq[3];
#pragma unroll
    for (int c = 0; c < 3; ++c) {
        const bf16_t* qp = Qb + (size_t)m16 * HD + c * 32 + hi * 8;
        aq[c].h[0] = *(const v8bf*)(qp);
        aq[c].h[1] = *(const v8bf*)(qp + 16);
    }

    v8f zero = {0.f,0.f,0.f,0.f,0.f,0.f,0.f,0.f};
    v8f acc[6];
#pragma unroll
    for (int t = 0; t < 6; ++t) acc[t] = zero;

    float run_max = -INFINITY, run_sum = 0.f;

    // strength-reduced row pointers
    const bf16_t* vp = Vb + m16;                   // += 32*HD per step
    const bf16_t* kp = Kb + m16;                   // += 32 per step

    // prologue: load step-0 fragments
    v16bf kf0[6], vf0[6], kf1[6], vf1[6];
#pragma unroll
    for (int t6 = 0; t6 < 6; ++t6)
#pragma unroll
        for (int s = 0; s < 16; ++s)
            vf0[t6][s] = vp[(size_t)(hi * 8 + (s & 7) + (s >> 3) * 16) * HD + 16 * t6];
#pragma unroll
    for (int t = 0; t < 2; ++t)
#pragma unroll
        for (int c = 0; c < 3; ++c)
#pragma unroll
            for (int s = 0; s < 16; ++s)
                kf0[t * 3 + c][s] = kp[(size_t)(c * 32 + hi * 8 + (s & 7) + (s >> 3) * 16) * SEQ + 16 * t];
    vp += (size_t)32 * HD;
    kp += 32;

    // 64 steps, 2x unrolled ping-pong (last step prefetches into padding)
    for (int it = 0; it < SEQ / 32; it += 2) {
        attn_step(sm, acc, aq, kf0, vf0, kf1, vf1, vp, kp,
                  run_max, run_sum, lane, m16, hi);
        attn_step(sm, acc, aq, kf1, vf1, kf0, vf0, vp, kp,
                  run_max, run_sum, lane, m16, hi);
    }

    if (lane < 16) sm.sum[lane] = run_sum;
    __syncthreads();

    float inv[8];
#pragma unroll
    for (int r = 0; r < 8; ++r) inv[r] = 1.f / (sm.sum[r + 8 * hi] * SCALE_F);

#pragma unroll
    for (int t6 = 0; t6 < 6; ++t6)
#pragma unroll
        for (int r = 0; r < 8; ++r) {
            int n = q0 + r + 8 * hi;
            int col = h * HD + 16 * t6 + m16;
            Out[((size_t)b * SEQ + n) * EMB + col] = (bf16_t)(acc[t6][r] * inv[r]);
        }
}

// ---------------------------------------------------------------------------
// LayerNorm(z)*g + b + residual -> f32 (optional) and bf16 (optional)
// one 256-thread block per row of 768
// ---------------------------------------------------------------------------
__global__ __launch_bounds__(256)
void layernorm_res_kernel(const float* __restrict__ Z, const float* __restrict__ g,
                          const float* __restrict__ bta, const float* __restrict__ res,
                          float* __restrict__ outf, bf16_t* __restrict__ outb) {
    __shared__ float red[256];
    const int row = blockIdx.x, tid = threadIdx.x;
    const float* z = Z + (size_t)row * EMB;

    float s = 0.f;
    for (int c = tid; c < EMB; c += 256) s += z[c];
    red[tid] = s; __syncthreads();
    for (int off = 128; off > 0; off >>= 1) {
        if (tid < off) red[tid] += red[tid + off];
        __syncthreads();
    }
    float mean = red[0] * (1.0f / EMB);
    __syncthreads();

    float v = 0.f;
    for (int c = tid; c < EMB; c += 256) { float d = z[c] - mean; v += d * d; }
    red[tid] = v; __syncthreads();
    for (int off = 128; off > 0; off >>= 1) {
        if (tid < off) red[tid] += red[tid + off];
        __syncthreads();
    }
    float rstd = rsqrtf(red[0] * (1.0f / EMB) + LN_EPS);

    for (int c = tid; c < EMB; c += 256) {
        float o = (z[c] - mean) * rstd * g[c] + bta[c] + res[(size_t)row * EMB + c];
        if (outf) outf[(size_t)row * EMB + c] = o;
        if (outb) outb[(size_t)row * EMB + c] = (bf16_t)o;
    }
}

// ---------------------------------------------------------------------------
// Final projection: [M,768] f32 x [768,2] + bias -> d_out [M,2] f32
// ---------------------------------------------------------------------------
__global__ __launch_bounds__(256)
void final_proj_kernel(const float* __restrict__ Y, const float* __restrict__ Wp,
                       const float* __restrict__ bp, float* __restrict__ out) {
    int row = blockIdx.x * 256 + threadIdx.x;
    if (row >= MROWS) return;
    const float* y = Y + (size_t)row * EMB;
    float a0 = 0.f, a1 = 0.f;
    for (int e = 0; e < EMB; ++e) {
        float v = y[e];
        a0 += v * Wp[e * 2 + 0];
        a1 += v * Wp[e * 2 + 1];
    }
    out[row * 2 + 0] = a0 + bp[0];
    out[row * 2 + 1] = a1 + bp[1];
}

// ---------------------------------------------------------------------------
extern "C" void kernel_launch(void* const* d_in, const int* in_sizes, int n_in,
                              void* d_out, int out_size, void* d_ws, size_t ws_size,
                              hipStream_t stream) {
    const float* x1      = (const float*)d_in[0];
    const float* x2      = (const float*)d_in[1];
    const float* ln_g    = (const float*)d_in[2];
    const float* ln_b    = (const float*)d_in[3];
    const float* W_mqkv  = (const float*)d_in[4];
    const float* b_mqkv  = (const float*)d_in[5];
    const float* W_mproj = (const float*)d_in[6];
    const float* b_mproj = (const float*)d_in[7];
    const float* W_qkv   = (const float*)d_in[8];
    const float* b_qkv   = (const float*)d_in[9];
    const float* W_bott  = (const float*)d_in[10];
    const float* b_bott  = (const float*)d_in[11];
    const float* W_proj  = (const float*)d_in[12];
    const float* b_proj  = (const float*)d_in[13];

    char* ws = (char*)d_ws;
    size_t off = 0;
    auto take = [&](size_t bytes) -> void* {
        void* p = ws + off;
        off += (bytes + 255) & ~(size_t)255;
        return p;
    };

    const size_t ME  = (size_t)MROWS * EMB;
    bf16_t* wm_bf  = (bf16_t*)take((size_t)EMB * NQKV * 2);
    bf16_t* wp_bf  = (bf16_t*)take((size_t)EMB * EMB * 2);
    bf16_t* wq_bf  = (bf16_t*)take((size_t)EMB * NQKV * 2);
    bf16_t* wb_bf  = (bf16_t*)take((size_t)EMB * EMB * 2);
    bf16_t* xbf    = (bf16_t*)take(ME * 2);
    float*  qkvf   = (float*) take((size_t)MROWS * NQKV * 4);
    bf16_t* Qb     = (bf16_t*)take(ME * 2);
    bf16_t* Ktb    = (bf16_t*)take(ME * 2 + 8192);   // +pad for phantom prefetch
    bf16_t* Vb     = (bf16_t*)take(ME * 2 + 8192);   // +pad for phantom prefetch
    bf16_t* attbf  = (bf16_t*)take(ME * 2);
    float*  gout   = (float*) take(ME * 4);
    float*  x1a_f  = (float*) take(ME * 4);
    float*  x2a_f  = (float*) take(ME * 4);
    bf16_t* x1a_b  = (bf16_t*)take(ME * 2);
    bf16_t* x2a_b  = (bf16_t*)take(ME * 2);
    float*  y_f    = (float*) take(ME * 4);

    auto cvt = [&](const float* s, bf16_t* d, size_t n) {
        cvt_f32_bf16_kernel<<<dim3((unsigned)((n + 255) / 256)), 256, 0, stream>>>(s, d, (int)n);
    };
    auto gemm = [&](const bf16_t* A, const bf16_t* Bw, const float* bias, float* C, int Nc) {
        gemm_bf16_wmma_kernel<<<dim3(MROWS / 64, Nc / 128), 256, 0, stream>>>(A, Bw, bias, C, EMB, Nc);
    };
    auto split = [&](int wq, int wk, int wv) {
        split_qkv_kernel<<<dim3((MROWS * EMB + 255) / 256), 256, 0, stream>>>(qkvf, Qb, Ktb, Vb, wq, wk, wv);
    };
    auto attn = [&]() {
        attention_kernel<<<dim3(BATCH * NH * (SEQ / 128)), 256, 0, stream>>>(Qb, Ktb, Vb, attbf);
    };

    // weight conversions
    cvt(W_mqkv,  wm_bf, (size_t)EMB * NQKV);
    cvt(W_mproj, wp_bf, (size_t)EMB * EMB);
    cvt(W_qkv,   wq_bf, (size_t)EMB * NQKV);
    cvt(W_bott,  wb_bf, (size_t)EMB * EMB);

    // ---- inner MHA, stream 1 ----
    cvt(x1, xbf, ME);
    gemm(xbf, wm_bf, b_mqkv, qkvf, NQKV);
    split(1, 1, 1);
    attn();
    gemm(attbf, wp_bf, b_mproj, gout, EMB);
    layernorm_res_kernel<<<dim3(MROWS), 256, 0, stream>>>(gout, ln_g, ln_b, x1, x1a_f, x1a_b);

    // ---- inner MHA, stream 2 ----
    cvt(x2, xbf, ME);
    gemm(xbf, wm_bf, b_mqkv, qkvf, NQKV);
    split(1, 1, 1);
    attn();
    gemm(attbf, wp_bf, b_mproj, gout, EMB);
    layernorm_res_kernel<<<dim3(MROWS), 256, 0, stream>>>(gout, ln_g, ln_b, x2, x2a_f, x2a_b);

    // ---- cross attention: q,k from stream 1, v from stream 2 ----
    gemm(x1a_b, wq_bf, b_qkv, qkvf, NQKV);
    split(1, 1, 0);                     // Q, Kt from x1_a
    gemm(x2a_b, wq_bf, b_qkv, qkvf, NQKV);
    split(0, 0, 1);                     // V from x2_a
    attn();
    gemm(attbf, wb_bf, b_bott, gout, EMB);
    layernorm_res_kernel<<<dim3(MROWS), 256, 0, stream>>>(gout, ln_g, ln_b, x2a_f, y_f, (bf16_t*)nullptr);

    // ---- final projection to 2 ----
    final_proj_kernel<<<dim3(MROWS / 256), 256, 0, stream>>>(y_f, W_proj, b_proj, (float*)d_out);
}
